// HopfieldPoolingRouter_18107582120392
// MI455X (gfx1250) — matrix operand order repaired and checked by reading, not compiled, not run on hardware
//
#include <hip/hip_runtime.h>
#include <cstdint>

#define H_NUM 16
#define DH 48
#define SCALING 0.25f
#define STEPS 5
#define LN_EPS 1e-5f
#define NEG_INF (-3.402823466e38f)

typedef __attribute__((ext_vector_type(16))) __bf16 v16bf;
typedef __attribute__((ext_vector_type(8)))  float  v8f;

__device__ __forceinline__ unsigned short f2bfbits(float f) {
  unsigned int u = __float_as_uint(f);
  return (unsigned short)((u + 0x7fffu + ((u >> 16) & 1u)) >> 16);
}
__device__ __forceinline__ unsigned int pack2bf(float a, float b) {
  return (unsigned int)f2bfbits(a) | ((unsigned int)f2bfbits(b) << 16);
}
__device__ __forceinline__ float bflo(unsigned int u) { return __uint_as_float(u << 16); }
__device__ __forceinline__ float bfhi(unsigned int u) { return __uint_as_float(u & 0xffff0000u); }

// ---------------- per-row LN stats + row L2 norm ----------------
__global__ __launch_bounds__(256) void k_lnstats(
    const float* __restrict__ mem, float* __restrict__ meanb,
    float* __restrict__ rstdb, float* __restrict__ rninv, int N)
{
  const int tid = threadIdx.x;
  const int lane = tid & 31, w = tid >> 5;
  int row = blockIdx.x * 8 + w;
  if (row >= N) return;
  const float* r = mem + (size_t)row * 768;
  float s = 0.f, ss = 0.f;
  for (int i = lane; i < 768; i += 32) { float x = r[i]; s += x; ss += x * x; }
  #pragma unroll
  for (int m = 16; m >= 1; m >>= 1) { s += __shfl_xor(s, m, 32); ss += __shfl_xor(ss, m, 32); }
  if (lane == 0) {
    float mean = s * (1.0f / 768.0f);
    float var = ss * (1.0f / 768.0f) - mean * mean;
    meanb[row] = mean;
    rstdb[row] = rsqrtf(var + LN_EPS);
    rninv[row] = rsqrtf(ss);
  }
}

// ---------------- fold LN gains into weights ----------------
// wtk: bf16, transposed (col-major over d) for contiguous B fragments
__global__ __launch_bounds__(256) void k_foldw(
    const float* __restrict__ Wk, const float* __restrict__ Wv,
    const float* __restrict__ gs, const float* __restrict__ gp,
    unsigned short* __restrict__ wtk, float* __restrict__ wvf)
{
  int idx = blockIdx.x * 256 + threadIdx.x;
  if (idx >= 768 * 768) return;
  int d = idx / 768, col = idx % 768;
  wtk[(size_t)col * 768 + d] = f2bfbits(gs[d] * Wk[idx]);
  wvf[idx] = gp[d] * Wv[idx];
}

// b'_k = bk + b_s @ Wk ; b'_v = bv + b_p @ Wv
__global__ __launch_bounds__(256) void k_foldb(
    const float* __restrict__ Wk, const float* __restrict__ Wv,
    const float* __restrict__ bs, const float* __restrict__ bp,
    const float* __restrict__ bk, const float* __restrict__ bv,
    float* __restrict__ bpk, float* __restrict__ bpv)
{
  int idx = blockIdx.x * 256 + threadIdx.x;
  if (idx >= 1536) return;
  int which = idx / 768, col = idx % 768;
  const float* W = which ? Wv : Wk;
  const float* bl = which ? bp : bs;
  float acc = 0.f;
  for (int d = 0; d < 768; ++d) acc += bl[d] * W[(size_t)d * 768 + col];
  if (which) bpv[col] = bv[col] + acc; else bpk[col] = bk[col] + acc;
}

// ---------------- WMMA GEMM: k = LN(memory) @ W'_k + b'_k (bf16 out) ----------------
__global__ __launch_bounds__(256) void k_gemm_wmma(
    const float* __restrict__ mem, const float* __restrict__ meanb,
    const float* __restrict__ rstdb, const unsigned short* __restrict__ wtk,
    const float* __restrict__ bpk, unsigned short* __restrict__ kout, int N)
{
  __shared__ unsigned short Alds[128 * 48];   // padded stride: 96B rows, 32B-aligned frags
  const int tid  = threadIdx.x;
  const int lane = tid & 31;
  const int wv   = tid >> 5;      // 8 waves
  const int wm   = wv >> 1;       // 0..3 (M)
  const int wn   = wv & 1;        // 0..1 (N)
  const int l15  = lane & 15;
  const int lhi  = lane >> 4;
  const int rowBlk = blockIdx.x * 128;
  const int colBlk = blockIdx.y * 64;

  // staging assignment: fixed row per thread (2 threads/row, 16 contiguous cols each)
  const int srow = tid >> 1;            // 0..127
  const int scol = (tid & 1) * 16;      // 0 or 16
  const int growS = rowBlk + srow;
  const bool svalid = growS < N;
  float smean = 0.f, srstd = 0.f;
  if (svalid) { smean = meanb[growS]; srstd = rstdb[growS]; }
  const float* srcRow = mem + (size_t)growS * 768 + scol;
  unsigned short* ldsDst = &Alds[srow * 48 + scol];

  v8f acc[2][2] = {};

  for (int kc = 0; kc < 768; kc += 32) {
    // stage A: 16 contiguous elements per thread, normalized + packed bf16
    uint4 u0 = {0u, 0u, 0u, 0u}, u1 = {0u, 0u, 0u, 0u};
    if (svalid) {
      const float4* s4 = (const float4*)(srcRow + kc);
      float4 f0 = s4[0], f1 = s4[1], f2 = s4[2], f3 = s4[3];
      u0.x = pack2bf((f0.x - smean) * srstd, (f0.y - smean) * srstd);
      u0.y = pack2bf((f0.z - smean) * srstd, (f0.w - smean) * srstd);
      u0.z = pack2bf((f1.x - smean) * srstd, (f1.y - smean) * srstd);
      u0.w = pack2bf((f1.z - smean) * srstd, (f1.w - smean) * srstd);
      u1.x = pack2bf((f2.x - smean) * srstd, (f2.y - smean) * srstd);
      u1.y = pack2bf((f2.z - smean) * srstd, (f2.w - smean) * srstd);
      u1.z = pack2bf((f3.x - smean) * srstd, (f3.y - smean) * srstd);
      u1.w = pack2bf((f3.z - smean) * srstd, (f3.w - smean) * srstd);
    }
    *(uint4*)ldsDst       = u0;
    *(uint4*)(ldsDst + 8) = u1;
    __syncthreads();

    // A 16x32 bf16 layout: lanes 0-15 hold K{0..15}; lanes 16-31 hold K{16..31}
    const int abase = lhi * 16;
    union { v16bf v; uint4 u4[2]; } af[2];
    v16bf bf[2];
    #pragma unroll
    for (int i = 0; i < 2; ++i) {
      int r = wm * 32 + i * 16 + l15;
      af[i].u4[0] = *(const uint4*)&Alds[r * 48 + abase];
      af[i].u4[1] = *(const uint4*)&Alds[r * 48 + abase + 8];
    }
    // B 32x16: lane column, K-contiguous halves; wtk is col-major so this is one 32B load
    #pragma unroll
    for (int j = 0; j < 2; ++j) {
      int col = colBlk + wn * 32 + j * 16 + l15;
      bf[j] = *(const v16bf*)&wtk[(size_t)col * 768 + kc + lhi * 16];
    }
    #pragma unroll
    for (int i = 0; i < 2; ++i)
      #pragma unroll
      for (int j = 0; j < 2; ++j)
        acc[i][j] = __builtin_amdgcn_wmma_f32_16x16x32_bf16(
            false, af[i].v, false, bf[j], (short)0, acc[i][j], false, false);
    __syncthreads();
  }

  #pragma unroll
  for (int i = 0; i < 2; ++i) {
    #pragma unroll
    for (int j = 0; j < 2; ++j) {
      int col = colBlk + wn * 32 + j * 16 + l15;
      float bias = bpk[col];
      int row0 = rowBlk + wm * 32 + i * 16 + lhi * 8;
      #pragma unroll
      for (int r = 0; r < 8; ++r) {
        int row = row0 + r;
        if (row < N)
          kout[(size_t)row * 768 + col] = f2bfbits(acc[i][j][r] + bias);
      }
    }
  }
}

// ---------------- q0 = 0.25 * (LN(query) @ Wq + bq) ----------------
__global__ __launch_bounds__(256) void k_qinit(
    const float* __restrict__ query, const float* __restrict__ g,
    const float* __restrict__ bln, const float* __restrict__ Wq,
    const float* __restrict__ bq, float* __restrict__ qv)
{
  __shared__ float xn[768];
  __shared__ float red[256];
  __shared__ float stat[2];
  const int tid = threadIdx.x;
  const int b = blockIdx.x;
  const float* qr = query + (size_t)b * 768;
  float s = 0.f, ss = 0.f;
  for (int i = tid; i < 768; i += 256) { float x = qr[i]; s += x; ss += x * x; }
  red[tid] = s; __syncthreads();
  for (int off = 128; off >= 1; off >>= 1) { if (tid < off) red[tid] += red[tid + off]; __syncthreads(); }
  if (tid == 0) stat[0] = red[0] * (1.0f / 768.0f);
  __syncthreads();
  red[tid] = ss; __syncthreads();
  for (int off = 128; off >= 1; off >>= 1) { if (tid < off) red[tid] += red[tid + off]; __syncthreads(); }
  if (tid == 0) {
    float mean = stat[0];
    float var = red[0] * (1.0f / 768.0f) - mean * mean;
    stat[1] = rsqrtf(var + LN_EPS);
  }
  __syncthreads();
  for (int i = tid; i < 768; i += 256)
    xn[i] = (qr[i] - stat[0]) * stat[1] * g[i] + bln[i];
  __syncthreads();
  for (int j = tid; j < 768; j += 256) {
    float acc = bq[j];
    for (int d = 0; d < 768; ++d) acc += xn[d] * Wq[(size_t)d * 768 + j];
    qv[(size_t)b * 768 + j] = SCALING * acc;
  }
}

// ---------------- logits + block max ----------------
__global__ __launch_bounds__(256) void k_logits_max(
    const unsigned short* __restrict__ kbuf, const float* __restrict__ qv,
    float* __restrict__ logits, float* __restrict__ pmax, int N, int nblk)
{
  __shared__ float qh[2][48];
  __shared__ float red[256];
  const int tid = threadIdx.x;
  const int h = blockIdx.y;
  if (tid < 96) { int b = tid / 48, j = tid % 48; qh[b][j] = qv[(size_t)(b * 16 + h) * 48 + j]; }
  __syncthreads();
  int n = blockIdx.x * 256 + tid;
  float a0 = NEG_INF, a1 = NEG_INF;
  if (n < N) {
    const uint4* kr = (const uint4*)(kbuf + (size_t)n * 768 + h * 48);
    float s0 = 0.f, s1 = 0.f;
    #pragma unroll
    for (int q4 = 0; q4 < 6; ++q4) {
      uint4 u = kr[q4];
      int j = q4 * 8;
      float f;
      f = bflo(u.x); s0 += qh[0][j+0]*f; s1 += qh[1][j+0]*f;
      f = bfhi(u.x); s0 += qh[0][j+1]*f; s1 += qh[1][j+1]*f;
      f = bflo(u.y); s0 += qh[0][j+2]*f; s1 += qh[1][j+2]*f;
      f = bfhi(u.y); s0 += qh[0][j+3]*f; s1 += qh[1][j+3]*f;
      f = bflo(u.z); s0 += qh[0][j+4]*f; s1 += qh[1][j+4]*f;
      f = bfhi(u.z); s0 += qh[0][j+5]*f; s1 += qh[1][j+5]*f;
      f = bflo(u.w); s0 += qh[0][j+6]*f; s1 += qh[1][j+6]*f;
      f = bfhi(u.w); s0 += qh[0][j+7]*f; s1 += qh[1][j+7]*f;
    }
    logits[(size_t)h * N + n] = s0;
    logits[(size_t)(16 + h) * N + n] = s1;
    a0 = s0; a1 = s1;
  }
  red[tid] = a0; __syncthreads();
  for (int off = 128; off >= 1; off >>= 1) { if (tid < off) red[tid] = fmaxf(red[tid], red[tid + off]); __syncthreads(); }
  if (tid == 0) pmax[(size_t)h * nblk + blockIdx.x] = red[0];
  __syncthreads();
  red[tid] = a1; __syncthreads();
  for (int off = 128; off >= 1; off >>= 1) { if (tid < off) red[tid] = fmaxf(red[tid], red[tid + off]); __syncthreads(); }
  if (tid == 0) pmax[(size_t)(16 + h) * nblk + blockIdx.x] = red[0];
}

// ---------------- per-(b,h) reduce: max or sum ----------------
__global__ __launch_bounds__(256) void k_reduce_bh(
    const float* __restrict__ part, float* __restrict__ outv, int cnt, int isMax)
{
  __shared__ float red[256];
  int bh = blockIdx.x, tid = threadIdx.x;
  float a = isMax ? NEG_INF : 0.f;
  for (int i = tid; i < cnt; i += 256) {
    float v = part[(size_t)bh * cnt + i];
    a = isMax ? fmaxf(a, v) : (a + v);
  }
  red[tid] = a; __syncthreads();
  for (int off = 128; off >= 1; off >>= 1) {
    if (tid < off) red[tid] = isMax ? fmaxf(red[tid], red[tid + off]) : (red[tid] + red[tid + off]);
    __syncthreads();
  }
  if (tid == 0) outv[bh] = red[0];
}

// ---------------- exp(logit - max) in place + block sums ----------------
__global__ __launch_bounds__(256) void k_expsum(
    float* __restrict__ logits, const float* __restrict__ gmax,
    float* __restrict__ psum, int N, int nblk)
{
  __shared__ float red[256];
  const int tid = threadIdx.x;
  const int h = blockIdx.y;
  int n = blockIdx.x * 256 + tid;
  float e0 = 0.f, e1 = 0.f;
  float gm0 = gmax[h], gm1 = gmax[16 + h];
  if (n < N) {
    size_t i0 = (size_t)h * N + n, i1 = (size_t)(16 + h) * N + n;
    e0 = __expf(logits[i0] - gm0);
    e1 = __expf(logits[i1] - gm1);
    logits[i0] = e0; logits[i1] = e1;
  }
  red[tid] = e0; __syncthreads();
  for (int off = 128; off >= 1; off >>= 1) { if (tid < off) red[tid] += red[tid + off]; __syncthreads(); }
  if (tid == 0) psum[(size_t)h * nblk + blockIdx.x] = red[0];
  __syncthreads();
  red[tid] = e1; __syncthreads();
  for (int off = 128; off >= 1; off >>= 1) { if (tid < off) red[tid] += red[tid + off]; __syncthreads(); }
  if (tid == 0) psum[(size_t)(16 + h) * nblk + blockIdx.x] = red[0];
}

// ---------------- q-update partials: qnew ∝ Σ p·k ----------------
__global__ __launch_bounds__(256) void k_qupd(
    const unsigned short* __restrict__ kbuf, const float* __restrict__ p,
    float* __restrict__ qpart, int N)
{
  __shared__ float wsum[8][48];
  const int tid = threadIdx.x;
  const int lane = tid & 31;
  const int w = tid >> 5;
  const int h = blockIdx.y;
  const int b = blockIdx.z;
  const int bh = b * 16 + h;
  const int base = blockIdx.x * 1024;
  float acc[48];
  #pragma unroll
  for (int j = 0; j < 48; ++j) acc[j] = 0.f;
  for (int t = 0; t < 4; ++t) {
    int n = base + t * 256 + tid;
    if (n < N) {
      float pv = p[(size_t)bh * N + n];
      const uint4* kr = (const uint4*)(kbuf + (size_t)n * 768 + h * 48);
      #pragma unroll
      for (int q4 = 0; q4 < 6; ++q4) {
        uint4 u = kr[q4];
        int j = q4 * 8;
        acc[j+0] += pv * bflo(u.x); acc[j+1] += pv * bfhi(u.x);
        acc[j+2] += pv * bflo(u.y); acc[j+3] += pv * bfhi(u.y);
        acc[j+4] += pv * bflo(u.z); acc[j+5] += pv * bfhi(u.z);
        acc[j+6] += pv * bflo(u.w); acc[j+7] += pv * bfhi(u.w);
      }
    }
  }
  #pragma unroll
  for (int j = 0; j < 48; ++j) {
    #pragma unroll
    for (int m = 16; m >= 1; m >>= 1) acc[j] += __shfl_xor(acc[j], m, 32);
  }
  if (lane == 0) {
    #pragma unroll
    for (int j = 0; j < 48; ++j) wsum[w][j] = acc[j];
  }
  __syncthreads();
  if (tid < 48) {
    float s = 0.f;
    #pragma unroll
    for (int ww = 0; ww < 8; ++ww) s += wsum[ww][tid];
    qpart[((size_t)blockIdx.x * 32 + bh) * 48 + tid] = s;
  }
}

__global__ __launch_bounds__(256) void k_qred(
    const float* __restrict__ qpart, const float* __restrict__ gsum,
    float* __restrict__ qv, int nchunk)
{
  int idx = blockIdx.x * 256 + threadIdx.x;
  if (idx >= 32 * 48) return;
  int bh = idx / 48, j = idx % 48;
  float s = 0.f;
  for (int c = 0; c < nchunk; ++c) s += qpart[((size_t)c * 32 + bh) * 48 + j];
  qv[idx] = SCALING * s / gsum[bh];
}

// ---------------- sv = Σ_n p[bh,n] * x_norm[n,:] (partials) ----------------
__global__ __launch_bounds__(128) void k_svpart(
    const float* __restrict__ mem, const float* __restrict__ meanb,
    const float* __restrict__ rstdb, const float* __restrict__ p,
    float* __restrict__ svpart, int N)
{
  __shared__ float pp[32][128];
  const int tid = threadIdx.x;
  const int d = blockIdx.y * 128 + tid;
  const int base = blockIdx.x * 1024;
  float acc[32];
  #pragma unroll
  for (int i = 0; i < 32; ++i) acc[i] = 0.f;
  for (int c0 = 0; c0 < 1024; c0 += 128) {
    __syncthreads();
    #pragma unroll
    for (int t = 0; t < 32; ++t) {
      int idx = t * 128 + tid;
      int bh = idx >> 7, r = idx & 127;
      int n = base + c0 + r;
      pp[bh][r] = (n < N) ? p[(size_t)bh * N + n] : 0.f;
    }
    __syncthreads();
    int rows = N - (base + c0);
    if (rows > 128) rows = 128;
    for (int r = 0; r < rows; ++r) {
      int n = base + c0 + r;
      float xn = (mem[(size_t)n * 768 + d] - meanb[n]) * rstdb[n];
      #pragma unroll
      for (int bh = 0; bh < 32; ++bh) acc[bh] += pp[bh][r] * xn;
    }
  }
  #pragma unroll
  for (int bh = 0; bh < 32; ++bh)
    svpart[((size_t)blockIdx.x * 32 + bh) * 768 + d] = acc[bh];
}

__global__ __launch_bounds__(256) void k_svred(
    const float* __restrict__ svpart, float* __restrict__ svb, int nchunk)
{
  int idx = blockIdx.x * 256 + threadIdx.x;
  if (idx >= 32 * 768) return;
  int bh = idx / 768, d = idx % 768;
  float s = 0.f;
  for (int c = 0; c < nchunk; ++c) s += svpart[((size_t)c * 32 + bh) * 768 + d];
  svb[idx] = s;
}

// ---------------- out768[b,h*48+j] = (sv/gsum) @ W'_v + b'_v ----------------
__global__ __launch_bounds__(64) void k_projv(
    const float* __restrict__ svb, const float* __restrict__ wvf,
    const float* __restrict__ bpv, const float* __restrict__ gsum,
    float* __restrict__ out768)
{
  int bh = blockIdx.x;
  int j = threadIdx.x;
  if (j >= 48) return;
  int b = bh >> 4, h = bh & 15;
  int col = h * 48 + j;
  float inv = 1.0f / gsum[bh];
  float acc = 0.f;
  for (int d = 0; d < 768; ++d) acc += svb[(size_t)bh * 768 + d] * wvf[(size_t)d * 768 + col];
  out768[(size_t)b * 768 + col] = acc * inv + bpv[col];
}

// ---------------- pooled = out768 @ Wo + bo → d_out[8..] ----------------
__global__ __launch_bounds__(256) void k_pooled(
    const float* __restrict__ out768, const float* __restrict__ Wo,
    const float* __restrict__ bo, float* __restrict__ outp)
{
  int idx = blockIdx.x * 256 + threadIdx.x;
  if (idx >= 1536) return;
  int b = idx / 768, j = idx % 768;
  float acc = bo[j];
  for (int d = 0; d < 768; ++d) acc += out768[(size_t)b * 768 + d] * Wo[(size_t)d * 768 + j];
  outp[8 + idx] = acc;
}

__global__ __launch_bounds__(256) void k_pnorm(
    const float* __restrict__ outbuf, float* __restrict__ pnorm)
{
  __shared__ float red[256];
  __shared__ float rn;
  int b = blockIdx.x, tid = threadIdx.x;
  const float* pr = outbuf + 8 + (size_t)b * 768;
  float ss = 0.f;
  for (int i = tid; i < 768; i += 256) { float x = pr[i]; ss += x * x; }
  red[tid] = ss; __syncthreads();
  for (int off = 128; off >= 1; off >>= 1) { if (tid < off) red[tid] += red[tid + off]; __syncthreads(); }
  if (tid == 0) rn = rsqrtf(red[0]);
  __syncthreads();
  for (int i = tid; i < 768; i += 256) pnorm[(size_t)b * 768 + i] = pr[i] * rn;
}

// ---------------- cosine similarity + per-block top-4 ----------------
__global__ __launch_bounds__(256) void k_sims(
    const float* __restrict__ mem, const float* __restrict__ rninv,
    const float* __restrict__ pnorm, float* __restrict__ candv,
    int* __restrict__ candi, int N)
{
  __shared__ float pn0[768], pn1[768];
  __shared__ float redv[256];
  __shared__ int redi[256];
  __shared__ int chi_s;
  const int tid = threadIdx.x;
  for (int i = tid; i < 768; i += 256) { pn0[i] = pnorm[i]; pn1[i] = pnorm[768 + i]; }
  __syncthreads();
  int n = blockIdx.x * 256 + tid;
  float s0 = NEG_INF, s1 = NEG_INF;
  if (n < N) {
    const float4* mr = (const float4*)(mem + (size_t)n * 768);
    float a0 = 0.f, a1 = 0.f;
    for (int i = 0; i < 192; ++i) {
      float4 m = mr[i]; int j = 4 * i;
      a0 += m.x * pn0[j] + m.y * pn0[j+1] + m.z * pn0[j+2] + m.w * pn0[j+3];
      a1 += m.x * pn1[j] + m.y * pn1[j+1] + m.z * pn1[j+2] + m.w * pn1[j+3];
    }
    float ri = rninv[n];
    s0 = a0 * ri; s1 = a1 * ri;
  }
  for (int b = 0; b < 2; ++b) {
    float myv = b ? s1 : s0;
    int myi = (n < N) ? n : -1;
    for (int sel = 0; sel < 4; ++sel) {
      __syncthreads();
      redv[tid] = myv; redi[tid] = myi;
      __syncthreads();
      for (int off = 128; off >= 1; off >>= 1) {
        if (tid < off) {
          float ov = redv[tid + off]; int oi = redi[tid + off];
          float cv = redv[tid];       int ci = redi[tid];
          bool take = (oi >= 0) && (ci < 0 || ov > cv || (ov == cv && oi < ci));
          if (take) { redv[tid] = ov; redi[tid] = oi; }
        }
        __syncthreads();
      }
      if (tid == 0) {
        candv[((size_t)blockIdx.x * 2 + b) * 4 + sel] = redv[0];
        candi[((size_t)blockIdx.x * 2 + b) * 4 + sel] = redi[0];
        chi_s = redi[0];
      }
      __syncthreads();
      if (myi >= 0 && myi == chi_s) myv = NEG_INF;
    }
  }
}

// ---------------- final top-4 merge → d_out[0..7] (as float) ----------------
__global__ __launch_bounds__(256) void k_topk(
    const float* __restrict__ candv, const int* __restrict__ candi,
    int nblk, float* __restrict__ outp)
{
  __shared__ float redv[256];
  __shared__ int redi[256];
  __shared__ int ch[4];
  int tid = threadIdx.x;
  int nc = nblk * 4;
  for (int b = 0; b < 2; ++b) {
    for (int sel = 0; sel < 4; ++sel) {
      float mv = NEG_INF; int mi = -1;
      for (int i = tid; i < nc; i += 256) {
        int addr = ((i >> 2) * 8) + b * 4 + (i & 3);
        float v = candv[addr]; int ix = candi[addr];
        if (ix < 0) continue;
        bool dup = false;
        for (int t = 0; t < sel; ++t) if (ch[t] == ix) dup = true;
        if (dup) continue;
        if (mi < 0 || v > mv || (v == mv && ix < mi)) { mv = v; mi = ix; }
      }
      redv[tid] = mv; redi[tid] = mi;
      __syncthreads();
      for (int off = 128; off >= 1; off >>= 1) {
        if (tid < off) {
          float ov = redv[tid + off]; int oi = redi[tid + off];
          float cv = redv[tid];       int ci = redi[tid];
          bool take = (oi >= 0) && (ci < 0 || ov > cv || (ov == cv && oi < ci));
          if (take) { redv[tid] = ov; redi[tid] = oi; }
        }
        __syncthreads();
      }
      if (tid == 0) { ch[sel] = redi[0]; outp[b * 4 + sel] = (float)redi[0]; }
      __syncthreads();
    }
  }
}

extern "C" void kernel_launch(void* const* d_in, const int* in_sizes, int n_in,
                              void* d_out, int out_size, void* d_ws, size_t ws_size,
                              hipStream_t stream)
{
  const float* query  = (const float*)d_in[0];
  const float* mem    = (const float*)d_in[1];
  const float* ln_q_g = (const float*)d_in[2];
  const float* ln_q_b = (const float*)d_in[3];
  const float* ln_s_g = (const float*)d_in[4];
  const float* ln_s_b = (const float*)d_in[5];
  const float* ln_p_g = (const float*)d_in[6];
  const float* ln_p_b = (const float*)d_in[7];
  const float* Wq = (const float*)d_in[8];  const float* bq = (const float*)d_in[9];
  const float* Wk = (const float*)d_in[10]; const float* bk = (const float*)d_in[11];
  const float* Wv = (const float*)d_in[12]; const float* bv = (const float*)d_in[13];
  const float* Wo = (const float*)d_in[14]; const float* bo = (const float*)d_in[15];
  (void)n_in; (void)out_size; (void)ws_size;

  const int N = in_sizes[1] / 768;
  const int nblkL  = (N + 255) / 256;
  const int nchunk = (N + 1023) / 1024;

  char* ws = (char*)d_ws;
  size_t off = 0;
  auto alloc = [&](size_t bytes) -> char* {
    char* p = ws + off;
    off = (off + bytes + 255) & ~(size_t)255;
    return p;
  };
  unsigned short* kbuf = (unsigned short*)alloc((size_t)N * 768 * 2);   // k (bf16), L2-resident
  unsigned short* wtk  = (unsigned short*)alloc((size_t)768 * 768 * 2); // folded Wk, transposed, bf16
  float* wvf    = (float*)alloc((size_t)768 * 768 * 4);                 // folded Wv, f32
  float* bpk    = (float*)alloc(768 * 4);
  float* bpv    = (float*)alloc(768 * 4);
  float* meanb  = (float*)alloc((size_t)N * 4);
  float* rstdb  = (float*)alloc((size_t)N * 4);
  float* rninv  = (float*)alloc((size_t)N * 4);
  float* qv     = (float*)alloc(1536 * 4);
  float* logits = (float*)alloc((size_t)32 * N * 4);                    // doubles as p (in-place exp)
  float* pmax   = (float*)alloc((size_t)32 * nblkL * 4);
  float* psum   = (float*)alloc((size_t)32 * nblkL * 4);
  float* gmax   = (float*)alloc(32 * 4);
  float* gsum   = (float*)alloc(32 * 4);
  float* qpart  = (float*)alloc((size_t)nchunk * 32 * 48 * 4);
  float* svpart = (float*)alloc((size_t)nchunk * 32 * 768 * 4);
  float* svb    = (float*)alloc((size_t)32 * 768 * 4);
  float* out768 = (float*)alloc(1536 * 4);
  float* pnorm  = (float*)alloc(1536 * 4);
  float* candv  = (float*)alloc((size_t)nblkL * 8 * 4);
  int*   candi  = (int*)alloc((size_t)nblkL * 8 * 4);
  float* outp   = (float*)d_out;

  // prep
  k_lnstats<<<(N + 7) / 8, 256, 0, stream>>>(mem, meanb, rstdb, rninv, N);
  k_foldw<<<(768 * 768 + 255) / 256, 256, 0, stream>>>(Wk, Wv, ln_s_g, ln_p_g, wtk, wvf);
  k_foldb<<<6, 256, 0, stream>>>(Wk, Wv, ln_s_b, ln_p_b, bk, bv, bpk, bpv);

  // k projection (WMMA bf16)
  k_gemm_wmma<<<dim3((N + 127) / 128, 12), 256, 0, stream>>>(mem, meanb, rstdb, wtk, bpk, kbuf, N);

  // initial q
  k_qinit<<<2, 256, 0, stream>>>(query, ln_q_g, ln_q_b, Wq, bq, qv);

  // 6 softmaxes, 5 Hopfield q-updates (k stays in L2)
  for (int it = 0; it <= STEPS; ++it) {
    k_logits_max<<<dim3(nblkL, 16), 256, 0, stream>>>(kbuf, qv, logits, pmax, N, nblkL);
    k_reduce_bh<<<32, 256, 0, stream>>>(pmax, gmax, nblkL, 1);
    k_expsum<<<dim3(nblkL, 16), 256, 0, stream>>>(logits, gmax, psum, N, nblkL);
    k_reduce_bh<<<32, 256, 0, stream>>>(psum, gsum, nblkL, 0);
    if (it < STEPS) {
      k_qupd<<<dim3(nchunk, 16, 2), 256, 0, stream>>>(kbuf, logits, qpart, N);
      k_qred<<<6, 256, 0, stream>>>(qpart, gsum, qv, nchunk);
    }
  }

  // out = attn @ v without materializing v: sv = Σ p·x_norm, then tiny projection
  k_svpart<<<dim3(nchunk, 6), 128, 0, stream>>>(mem, meanb, rstdb, logits, svpart, N);
  k_svred<<<96, 256, 0, stream>>>(svpart, svb, nchunk);
  k_projv<<<32, 64, 0, stream>>>(svb, wvf, bpv, gsum, out768);
  k_pooled<<<6, 256, 0, stream>>>(out768, Wo, bo, outp);
  k_pnorm<<<2, 256, 0, stream>>>(outp, pnorm);

  // cosine similarity + top-4
  k_sims<<<nblkL, 256, 0, stream>>>(mem, rninv, pnorm, candv, candi, N);
  k_topk<<<1, 256, 0, stream>>>(candv, candi, nblkL, outp);
}
